// AttnBlock_88493506167032
// MI455X (gfx1250) — matrix operand order repaired and checked by reading, not compiled, not run on hardware
//
#include <hip/hip_runtime.h>
#include <math.h>

// ---------------------------------------------------------------------------
// CDNA5 (gfx1250) wave32 WMMA attention block.
//   x:[1,512,8,32,32] f32 -> GroupNorm -> q,k,v (1x1 conv) -> softmax(qT k) v
//   -> proj -> +x.   C=512, N=8192, 16 groups.
// All GEMMs use v_wmma_f32_16x16x32_bf16 (f32 accumulate). Attention is a
// flash-style streaming softmax so the 8192x8192 score matrix is never
// materialized. Intermediates (~8MB each, bf16) stay resident in the 192MB L2.
// ---------------------------------------------------------------------------

typedef __bf16 bf16_t;
typedef bf16_t bf16x16 __attribute__((ext_vector_type(16)));
typedef bf16_t bf16x8  __attribute__((ext_vector_type(8)));
typedef float  f32x8   __attribute__((ext_vector_type(8)));

#define C_DIM 512
#define N_TOK 8192
#define GROUP_ELEMS 262144   // 32 ch * 8192
#define LOG2E 1.4426950408889634f

static __device__ __forceinline__ f32x8 wmma_bf16(bf16x16 a, bf16x16 b, f32x8 c) {
  return __builtin_amdgcn_wmma_f32_16x16x32_bf16(false, a, false, b, (short)0, c,
                                                 false, false);
}

// A-matrix fragment (16x32 bf16, interleaved layout per ISA 7.12.2):
// lane L: row M = L%16, g = L/16; slots 0..7 -> K = 8g+0..7, slots 8..15 ->
// K = 16+8g+0..7.  'base' points at (row, Kbase + 8g).
static __device__ __forceinline__ bf16x16 load_frag_A(const bf16_t* base) {
  union { bf16x16 v; bf16x8 h[2]; } u;
  u.h[0] = *reinterpret_cast<const bf16x8*>(base);
  u.h[1] = *reinterpret_cast<const bf16x8*>(base + 16);
  return u.v;
}

// B-matrix fragment (32x16 bf16): lane L: col N = L%16, g = L/16;
// slots 0..7 -> K = 8g+0..7, slots 8..15 -> K = 16+8g+0..7.  'base' points at
// (Kbase+8g, col) in a K-contiguous-per-row layout, i.e. two 16B loads.
static __device__ __forceinline__ bf16x16 load_frag_B(const bf16_t* base) {
  union { bf16x16 v; bf16x8 h[2]; } u;
  u.h[0] = *reinterpret_cast<const bf16x8*>(base);
  u.h[1] = *reinterpret_cast<const bf16x8*>(base + 16);
  return u.v;
}

// -------------------------- small prep kernels -----------------------------

__global__ void f2bf_kernel(const float* __restrict__ src,
                            bf16_t* __restrict__ dst, int n) {
  int i = blockIdx.x * 256 + threadIdx.x;
  if (i < n) dst[i] = (bf16_t)src[i];
}

__global__ void gn_reduce_kernel(const float* __restrict__ x,
                                 float* __restrict__ psum,
                                 float* __restrict__ psq) {
  int g = blockIdx.y, ch = blockIdx.x;                 // 16 groups x 32 chunks
  const float* p = x + (size_t)g * GROUP_ELEMS + (size_t)ch * 8192;
  float s = 0.f, q = 0.f;
  for (int i = threadIdx.x; i < 8192; i += 256) { float v = p[i]; s += v; q += v * v; }
  __shared__ float ss[256], qq[256];
  ss[threadIdx.x] = s; qq[threadIdx.x] = q;
  __syncthreads();
  for (int off = 128; off > 0; off >>= 1) {
    if (threadIdx.x < off) { ss[threadIdx.x] += ss[threadIdx.x + off];
                             qq[threadIdx.x] += qq[threadIdx.x + off]; }
    __syncthreads();
  }
  if (threadIdx.x == 0) { psum[g * 32 + ch] = ss[0]; psq[g * 32 + ch] = qq[0]; }
}

__global__ void gn_stats_kernel(const float* __restrict__ psum,
                                const float* __restrict__ psq,
                                float* __restrict__ meanA,
                                float* __restrict__ rstdA) {
  int g = threadIdx.x;
  if (g < 16) {
    float s = 0.f, q = 0.f;
    for (int c = 0; c < 32; ++c) { s += psum[g * 32 + c]; q += psq[g * 32 + c]; }
    float mean = s * (1.f / (float)GROUP_ELEMS);
    float var  = q * (1.f / (float)GROUP_ELEMS) - mean * mean;
    meanA[g] = mean;
    rstdA[g] = rsqrtf(var + 1e-6f);
  }
}

__global__ void gn_norm_kernel(const float* __restrict__ x,
                               const float* __restrict__ gamma,
                               const float* __restrict__ beta,
                               const float* __restrict__ meanA,
                               const float* __restrict__ rstdA,
                               bf16_t* __restrict__ hnT) {
  int idx = blockIdx.x * 256 + threadIdx.x;            // 512*8192 elements
  int c = idx >> 13, n = idx & 8191, g = c >> 5;
  float v = (x[idx] - meanA[g]) * rstdA[g] * gamma[c] + beta[c];
  hnT[(size_t)n * C_DIM + c] = (bf16_t)v;              // transposed [n][c]
}

// ------------------------------ QKV GEMM -----------------------------------
// out = W(512x512) * hn(512x8192) + bias ; hn passed as hnT[n][c].
// transposed=1 -> write out as [n][c] (for qT/kT), else [c][n] (for v).
// 'scale' folds C^-0.5 into q.
__global__ void qkv_gemm_kernel(const bf16_t* __restrict__ W,
                                const float* __restrict__ bias,
                                const bf16_t* __restrict__ hnT,
                                bf16_t* __restrict__ out,
                                float scale, int transposed) {
  const int lane = threadIdx.x & 31;
  const int w    = threadIdx.x >> 5;
  const int n0   = blockIdx.x * 16;
  const int o0   = (blockIdx.y * 8 + w) * 16;
  const int M    = lane & 15;    // A-row / B-col / D-col
  const int g    = lane >> 4;

  f32x8 acc = {};
  for (int t = 0; t < 16; ++t) {
    bf16x16 A = load_frag_A(W   + (size_t)(o0 + M) * C_DIM + t * 32 + 8 * g);
    bf16x16 B = load_frag_B(hnT + (size_t)(n0 + M) * C_DIM + t * 32 + 8 * g);
    acc = wmma_bf16(A, B, acc);
  }
  // D layout: element(Mr,Nc): lane = (Mr/8)*16 + Nc, vgpr r = Mr%8.
  if (transposed) {
    bf16x8 pk;
    for (int r = 0; r < 8; ++r) {
      int o = o0 + r + 8 * g;
      pk[r] = (bf16_t)((acc[r] + bias[o]) * scale);
    }
    *reinterpret_cast<bf16x8*>(out + (size_t)(n0 + M) * C_DIM + o0 + 8 * g) = pk;
  } else {
    for (int r = 0; r < 8; ++r) {
      int o = o0 + r + 8 * g;
      out[(size_t)o * N_TOK + n0 + M] = (bf16_t)((acc[r] + bias[o]) * scale);
    }
  }
}

// --------------------------- flash attention -------------------------------
// Block: 32 query rows (n0..n0+31), 8 waves = 2 row-subtiles x 4 c-slices(128).
// Streams m in steps of 32 with online softmax. Emits hoT[n][c] bf16.
__global__ void attn_flash_kernel(const bf16_t* __restrict__ qT,
                                  const bf16_t* __restrict__ kT,
                                  const bf16_t* __restrict__ Vb,
                                  bf16_t* __restrict__ hoT) {
  const int tid  = threadIdx.x;
  const int lane = tid & 31;
  const int w    = tid >> 5;
  const int i    = w & 1;        // row-subtile (16 rows)
  const int j    = w >> 1;       // c-slice (128 channels)
  const int n0   = blockIdx.x * 32;
  const int M    = lane & 15;
  const int g    = lane >> 4;

  __shared__ float Spart[4][2][16][32];               // per-c-slice partial scores
  __shared__ __align__(32) bf16_t Pbuf[2][16][32];    // softmax probs (bf16)
  __shared__ float rowMax[32], rowSum[32], rowFac[32];

  if (tid < 32) { rowMax[tid] = -INFINITY; rowSum[tid] = 0.f; rowFac[tid] = 0.f; }

  // Preload this wave's qT A-fragments (16 rows x 128 c = 4 frags).
  bf16x16 Aq[4];
  for (int t = 0; t < 4; ++t)
    Aq[t] = load_frag_A(qT + (size_t)(n0 + i * 16 + M) * C_DIM + j * 128 + t * 32 + 8 * g);

  f32x8 acc[8];
  for (int u = 0; u < 8; ++u) acc[u] = f32x8{};
  __syncthreads();

  for (int m0 = 0; m0 < N_TOK; m0 += 32) {
    // ---- partial scores over this wave's 128-c slice ----
    for (int msub = 0; msub < 2; ++msub) {
      f32x8 s = {};
      for (int t = 0; t < 4; ++t) {
        bf16x16 B = load_frag_B(kT + (size_t)(m0 + msub * 16 + M) * C_DIM +
                                j * 128 + t * 32 + 8 * g);
        s = wmma_bf16(Aq[t], B, s);
      }
      for (int r = 0; r < 8; ++r)
        Spart[j][i][r + 8 * g][msub * 16 + M] = s[r];
    }
    __syncthreads();

    // ---- online softmax: wave w owns rows 4w..4w+3 ----
    for (int rr = 0; rr < 4; ++rr) {
      int row = w * 4 + rr;
      int ns = row >> 4, rl = row & 15;
      float s = Spart[0][ns][rl][lane] + Spart[1][ns][rl][lane] +
                Spart[2][ns][rl][lane] + Spart[3][ns][rl][lane];
      float mx = s;
      for (int off = 16; off > 0; off >>= 1) mx = fmaxf(mx, __shfl_xor(mx, off, 32));
      float mold = rowMax[row];
      float mnew = fmaxf(mold, mx);
      float p = exp2f((s - mnew) * LOG2E);
      float su = p;
      for (int off = 16; off > 0; off >>= 1) su += __shfl_xor(su, off, 32);
      Pbuf[ns][rl][lane] = (bf16_t)p;
      if (lane == 0) {
        float fac = exp2f((mold - mnew) * LOG2E);
        rowFac[row] = fac;
        rowSum[row] = rowSum[row] * fac + su;
        rowMax[row] = mnew;
      }
    }
    __syncthreads();

    // ---- P x V into this wave's 16n x 128c accumulators ----
    float fr[8];
    for (int r = 0; r < 8; ++r) fr[r] = rowFac[i * 16 + r + 8 * g];
    bf16x16 Ap = load_frag_A(&Pbuf[i][M][8 * g]);
    for (int u = 0; u < 8; ++u) {
      for (int r = 0; r < 8; ++r) acc[u][r] *= fr[r];
      bf16x16 B = load_frag_B(Vb + (size_t)(j * 128 + u * 16 + M) * N_TOK +
                              m0 + 8 * g);
      acc[u] = wmma_bf16(Ap, B, acc[u]);
    }
    __syncthreads();
  }

  // ---- finalize: divide by row sums, write hoT[n][c] ----
  float inv[8];
  for (int r = 0; r < 8; ++r) inv[r] = 1.f / rowSum[i * 16 + r + 8 * g];
  for (int u = 0; u < 8; ++u)
    for (int r = 0; r < 8; ++r)
      hoT[(size_t)(n0 + i * 16 + r + 8 * g) * C_DIM + j * 128 + u * 16 + M] =
          (bf16_t)(acc[u][r] * inv[r]);
}

// ------------------------ projection + residual ----------------------------
__global__ void proj_residual_kernel(const bf16_t* __restrict__ Wp,
                                     const float* __restrict__ bp,
                                     const bf16_t* __restrict__ hoT,
                                     const float* __restrict__ x,
                                     float* __restrict__ out) {
  const int lane = threadIdx.x & 31;
  const int w    = threadIdx.x >> 5;
  const int n0   = blockIdx.x * 16;
  const int o0   = (blockIdx.y * 8 + w) * 16;
  const int M    = lane & 15;
  const int g    = lane >> 4;

  f32x8 acc = {};
  for (int t = 0; t < 16; ++t) {
    bf16x16 A = load_frag_A(Wp  + (size_t)(o0 + M) * C_DIM + t * 32 + 8 * g);
    bf16x16 B = load_frag_B(hoT + (size_t)(n0 + M) * C_DIM + t * 32 + 8 * g);
    acc = wmma_bf16(A, B, acc);
  }
  for (int r = 0; r < 8; ++r) {
    int o = o0 + r + 8 * g;
    size_t idx = (size_t)o * N_TOK + n0 + M;
    out[idx] = acc[r] + bp[o] + x[idx];
  }
}

// ------------------------------- launch ------------------------------------
extern "C" void kernel_launch(void* const* d_in, const int* in_sizes, int n_in,
                              void* d_out, int out_size, void* d_ws, size_t ws_size,
                              hipStream_t stream) {
  (void)in_sizes; (void)n_in; (void)out_size;
  const float* x     = (const float*)d_in[0];
  const float* gamma = (const float*)d_in[1];
  const float* beta  = (const float*)d_in[2];
  const float* Wq    = (const float*)d_in[3];
  const float* bq    = (const float*)d_in[4];
  const float* Wk    = (const float*)d_in[5];
  const float* bk    = (const float*)d_in[6];
  const float* Wv    = (const float*)d_in[7];
  const float* bv    = (const float*)d_in[8];
  const float* Wp    = (const float*)d_in[9];
  const float* bp    = (const float*)d_in[10];
  float* out = (float*)d_out;

  // Workspace layout (needs ~44.1 MB; assumed <= ws_size).
  char* ws = (char*)d_ws;
  float* psum  = (float*)(ws + 0);          // 16*32
  float* psq   = (float*)(ws + 2048);       // 16*32
  float* meanA = (float*)(ws + 4096);       // 16
  float* rstdA = (float*)(ws + 4160);       // 16
  size_t off = 4352;                        // 256-aligned
  const size_t WB = (size_t)C_DIM * C_DIM * sizeof(bf16_t);   // 512KB
  bf16_t* Wqb = (bf16_t*)(ws + off); off += WB;
  bf16_t* Wkb = (bf16_t*)(ws + off); off += WB;
  bf16_t* Wvb = (bf16_t*)(ws + off); off += WB;
  bf16_t* Wpb = (bf16_t*)(ws + off); off += WB;
  const size_t TB = (size_t)N_TOK * C_DIM * sizeof(bf16_t);   // 8MB
  bf16_t* hnT  = (bf16_t*)(ws + off); off += TB;
  bf16_t* qTb  = (bf16_t*)(ws + off); off += TB;
  bf16_t* kTb  = (bf16_t*)(ws + off); off += TB;
  bf16_t* vBb  = (bf16_t*)(ws + off); off += TB;
  bf16_t* hoT  = (bf16_t*)(ws + off); off += TB;
  (void)ws_size;

  const int WN = C_DIM * C_DIM;                       // 262144
  f2bf_kernel<<<WN / 256, 256, 0, stream>>>(Wq, Wqb, WN);
  f2bf_kernel<<<WN / 256, 256, 0, stream>>>(Wk, Wkb, WN);
  f2bf_kernel<<<WN / 256, 256, 0, stream>>>(Wv, Wvb, WN);
  f2bf_kernel<<<WN / 256, 256, 0, stream>>>(Wp, Wpb, WN);

  gn_reduce_kernel<<<dim3(32, 16), 256, 0, stream>>>(x, psum, psq);
  gn_stats_kernel<<<1, 16, 0, stream>>>(psum, psq, meanA, rstdA);
  gn_norm_kernel<<<(C_DIM * N_TOK) / 256, 256, 0, stream>>>(x, gamma, beta,
                                                            meanA, rstdA, hnT);

  const float qscale = 0.044194173824159216f;  // 512^-0.5 folded into q
  qkv_gemm_kernel<<<dim3(512, 4), 256, 0, stream>>>(Wqb, bq, hnT, qTb, qscale, 1);
  qkv_gemm_kernel<<<dim3(512, 4), 256, 0, stream>>>(Wkb, bk, hnT, kTb, 1.0f, 1);
  qkv_gemm_kernel<<<dim3(512, 4), 256, 0, stream>>>(Wvb, bv, hnT, vBb, 1.0f, 0);

  attn_flash_kernel<<<N_TOK / 32, 256, 0, stream>>>(qTb, kTb, vBb, hoT);

  proj_residual_kernel<<<dim3(512, 4), 256, 0, stream>>>(Wpb, bp, hoT, x, out);
}
